// MultiHeadAttention_2765958938915
// MI455X (gfx1250) — matrix operand order repaired and checked by reading, not compiled, not run on hardware
//
#include <hip/hip_runtime.h>
#include <hip/hip_bf16.h>
#include <cstdint>
#include <cstddef>

// ---------------------------------------------------------------------------
// MI455X (gfx1250) multi-head attention, bf16 WMMA + Tensor Data Mover.
// B=4, S=2048, D=1024, H=16, HD=64.
//  - all matmuls via v_wmma_f32_16x16x32_bf16 (f32 accumulate)
//  - global->LDS tile staging via TDM (tensor_load_to_lds) with hardware
//    LDS row padding (+8 bf16 per row) for bank-conflict-free ds_load_b128
//  - double-buffered GEMM k-loop: one barrier per k-step, DMA prefetch
// Workspace: ~88 MB.
// ---------------------------------------------------------------------------

typedef __bf16 bf16_t;
typedef __attribute__((ext_vector_type(8)))  __bf16 v8bf;
typedef __attribute__((ext_vector_type(16))) __bf16 v16bf;
typedef __attribute__((ext_vector_type(8)))  float  v8f;
typedef __attribute__((ext_vector_type(4)))  unsigned int v4u;
typedef __attribute__((ext_vector_type(8)))  int v8i;
typedef __attribute__((ext_vector_type(4)))  int v4i;

#define WMMA_BF16(A, B, C) \
  __builtin_amdgcn_wmma_f32_16x16x32_bf16(false, (A), false, (B), (short)0, (C), false, false)

#if defined(__has_builtin)
#if __has_builtin(__builtin_amdgcn_tensor_load_to_lds) && \
    __has_builtin(__builtin_amdgcn_s_wait_tensorcnt)
#define HAVE_TDM 1
#endif
#endif
#ifndef HAVE_TDM
#define HAVE_TDM 0
#endif

static constexpr int BATCH  = 4;
static constexpr int SEQ    = 2048;
static constexpr int DMODEL = 1024;
static constexpr int NHEAD  = 16;
static constexpr int HDIM   = 64;
static constexpr float ATTN_SCALE = 0.125f;             // HD^-0.5
static constexpr float LOG2E = 1.44269504088896340736f;

// Build a 16-element bf16 fragment from two 8-element chunks at p[0..7] and
// p[gap..gap+7].  A-frag (ISA 7.12.2 16-bit A 16x32): gap=16.  B-frag: gap=16.
__device__ __forceinline__ v16bf make_frag(const bf16_t* p, int gap) {
  v8bf lo = *(const v8bf*)p;
  v8bf hi = *(const v8bf*)(p + gap);
  v16bf r;
#pragma unroll
  for (int i = 0; i < 8; ++i) { r[i] = lo[i]; r[i + 8] = hi[i]; }
  return r;
}

// Flat LDS pointer -> 32-bit LDS byte offset (aperture maps addr[31:0]).
__device__ __forceinline__ unsigned lds_off(const void* p) {
  return (unsigned)(uintptr_t)p;
}

#if HAVE_TDM
// 2D bf16 tile DMA: global (row stride `stride_e` elements) -> LDS, tile
// tw x th elements, LDS rows padded +8 bf16 (4 DWORDs) every 32 DWORDs
// (tw must be 64: pad_interval code 4 = 32 DW, pad_amount code 3 = 4 DW).
__device__ __forceinline__ void tdm_load_2d_bf16(
    unsigned lds_addr, const bf16_t* gptr, unsigned tw, unsigned th,
    unsigned stride_e) {
  unsigned long long ga = (unsigned long long)(uintptr_t)gptr;
  v4u g0;
  g0[0] = 1u;                                        // count=1, user mode
  g0[1] = lds_addr;                                  // LDS byte address
  g0[2] = (unsigned)(ga & 0xFFFFFFFFu);              // global_addr[31:0]
  g0[3] = (unsigned)((ga >> 32) & 0x1FFFFFFu) | (2u << 30);  // [56:32] | type=2
  v8i g1;
  const unsigned w0 = (1u << 16)                     // data_size = 2 bytes
                    | (1u << 20)                     // pad_enable
                    | (4u << 22)                     // pad_interval: 32 DW
                    | (3u << 25);                    // pad_amount:   4 DW
  g1[0] = (int)w0;
  g1[1] = (int)((tw & 0xFFFFu) << 16);               // tensor_dim0[15:0]
  g1[2] = (int)((tw >> 16) | ((th & 0xFFFFu) << 16));// dim0 hi | dim1 lo
  g1[3] = (int)((th >> 16) | (tw << 16));            // dim1 hi | tile_dim0
  g1[4] = (int)th;                                   // tile_dim1 (tile_dim2=0)
  g1[5] = (int)stride_e;                             // tensor_dim0_stride lo
  g1[6] = 0;                                         // stride0 hi | stride1 lo
  g1[7] = 0;
  v4i z4 = {0, 0, 0, 0};
  v8i z8 = {0, 0, 0, 0, 0, 0, 0, 0};
  // 6-arg toolchain signature: (g0, g1, g2, g3, extra, cpol)
  __builtin_amdgcn_tensor_load_to_lds(g0, g1, z4, z4, z8, 0);
}
#endif

// wave-uniform scalar condition (forces s_cbranch, not EXEC masking; TDM
// ignores EXEC so a masked lowering would issue duplicate DMAs per wave).
__device__ __forceinline__ bool is_wave0() {
  return __builtin_amdgcn_readfirstlane((int)(threadIdx.x >> 5)) == 0;
}

// ---------------------------------------------------------------------------
// fp32 -> bf16 conversion (8 elements / thread; sizes are multiples of 2048)
// ---------------------------------------------------------------------------
__global__ __launch_bounds__(256) void cvt_f32_to_bf16(
    const float* __restrict__ in, bf16_t* __restrict__ out, int n) {
  int i = (blockIdx.x * 256 + threadIdx.x) * 8;
  if (i + 8 <= n) {
    float4 a = *(const float4*)(in + i);
    float4 b = *(const float4*)(in + i + 4);
    v8bf o;
    o[0] = (__bf16)a.x; o[1] = (__bf16)a.y; o[2] = (__bf16)a.z; o[3] = (__bf16)a.w;
    o[4] = (__bf16)b.x; o[5] = (__bf16)b.y; o[6] = (__bf16)b.z; o[7] = (__bf16)b.w;
    *(v8bf*)(out + i) = o;
  }
}

// ---------------------------------------------------------------------------
// Tiled bf16 WMMA GEMM:  C[M,N] = A[M,K] @ W[N,K]^T + bias[N]
//   EPI == 0 : fp32 C (final projection)
//   EPI == 1 : scatter bias-added bf16 into Q/K/V [B,H,S,HD]
// 256 threads (8 wave32), BM=128 x BN=64, BK=64, TDM double buffering,
// one barrier per k-step, 8 wmma per wave per k-step.
// ---------------------------------------------------------------------------
template <int EPI>
__global__ __launch_bounds__(256) void gemm_bf16_wmma(
    const bf16_t* __restrict__ A, const bf16_t* __restrict__ W,
    const float* __restrict__ bias,
    float*  __restrict__ outF,
    bf16_t* __restrict__ qb, bf16_t* __restrict__ kb, bf16_t* __restrict__ vb,
    int M, int N, int Kdim) {
  constexpr int BM = 128, BN = 64, BK = 64, LDK = BK + 8;
  __shared__ bf16_t As[2][BM][LDK];   // 2 x 18 KB
  __shared__ bf16_t Bs[2][BN][LDK];   // 2 x  9 KB

  const int tid   = threadIdx.x;
  const int lane  = tid & 31;
  const int wave  = tid >> 5;
  const int ln    = lane & 15;
  const int half8 = (lane < 16) ? 0 : 8;
  const int m0 = blockIdx.x * BM;
  const int n0 = blockIdx.y * BN;
  const int wm = (wave >> 1) * 32;
  const int wn = (wave & 1) * 32;

  v8f acc[2][2];
#pragma unroll
  for (int i = 0; i < 2; ++i)
#pragma unroll
    for (int j = 0; j < 2; ++j)
#pragma unroll
      for (int r = 0; r < 8; ++r) acc[i][j][r] = 0.0f;

  const int nt = Kdim / BK;

#if HAVE_TDM
  if (is_wave0()) {
    tdm_load_2d_bf16(lds_off(&As[0][0][0]), A + (size_t)m0 * Kdim, BK, BM, Kdim);
    tdm_load_2d_bf16(lds_off(&Bs[0][0][0]), W + (size_t)n0 * Kdim, BK, BN, Kdim);
  }
#endif

  for (int t = 0; t < nt; ++t) {
    const int cb = t & 1;
#if HAVE_TDM
    if (is_wave0()) __builtin_amdgcn_s_wait_tensorcnt(0);
    __syncthreads();                       // tile t visible; buf t^1 free
    if (t + 1 < nt && is_wave0()) {
      const int k1 = (t + 1) * BK;
      tdm_load_2d_bf16(lds_off(&As[cb ^ 1][0][0]), A + (size_t)m0 * Kdim + k1, BK, BM, Kdim);
      tdm_load_2d_bf16(lds_off(&Bs[cb ^ 1][0][0]), W + (size_t)n0 * Kdim + k1, BK, BN, Kdim);
    }
#else
    {  // cooperative staging fallback (single logical buffer per step)
      const int k0 = t * BK;
      const int arow = tid >> 1, ac = (tid & 1) * 32;
      const bf16_t* src = A + (size_t)(m0 + arow) * Kdim + k0 + ac;
#pragma unroll
      for (int c = 0; c < 4; ++c)
        *(v8bf*)&As[cb][arow][ac + c * 8] = *(const v8bf*)(src + c * 8);
      if (tid < 128) {
        const int brow = tid >> 1, bc = (tid & 1) * 32;
        const bf16_t* srcb = W + (size_t)(n0 + brow) * Kdim + k0 + bc;
#pragma unroll
        for (int c = 0; c < 4; ++c)
          *(v8bf*)&Bs[cb][brow][bc + c * 8] = *(const v8bf*)(srcb + c * 8);
      }
      __syncthreads();
    }
#endif

#pragma unroll
    for (int ks = 0; ks < 2; ++ks) {
      v16bf af0 = make_frag(&As[cb][wm + ln][ks * 32 + half8], 16);
      v16bf af1 = make_frag(&As[cb][wm + 16 + ln][ks * 32 + half8], 16);
      v16bf bf0 = make_frag(&Bs[cb][wn + ln][ks * 32 + half8], 16);
      v16bf bf1 = make_frag(&Bs[cb][wn + 16 + ln][ks * 32 + half8], 16);
      acc[0][0] = WMMA_BF16(af0, bf0, acc[0][0]);
      acc[0][1] = WMMA_BF16(af0, bf1, acc[0][1]);
      acc[1][0] = WMMA_BF16(af1, bf0, acc[1][0]);
      acc[1][1] = WMMA_BF16(af1, bf1, acc[1][1]);
    }
#if !HAVE_TDM
    __syncthreads();
#endif
  }

  // Epilogue. C layout: VGPR r -> row (r + half8), column = ln per 16x16 tile.
#pragma unroll
  for (int i = 0; i < 2; ++i) {
    const int mbase = m0 + wm + i * 16 + half8;
#pragma unroll
    for (int j = 0; j < 2; ++j) {
      const int n = n0 + wn + j * 16 + ln;
      const float bv = bias[n];
      if constexpr (EPI == 0) {
#pragma unroll
        for (int r = 0; r < 8; ++r)
          outF[(size_t)(mbase + r) * N + n] = acc[i][j][r] + bv;
      } else {
        const int which = n >> 10;
        const int d  = n & 1023;
        const int h  = d >> 6;
        const int hd = d & 63;
        bf16_t* dst = (which == 0) ? qb : ((which == 1) ? kb : vb);
#pragma unroll
        for (int r = 0; r < 8; ++r) {
          const int m = mbase + r;
          const int b = m >> 11, s = m & 2047;
          dst[(((size_t)(b * NHEAD + h) * SEQ) + s) * HDIM + hd] =
              (bf16_t)(acc[i][j][r] + bv);
        }
      }
    }
  }
}

// ---------------------------------------------------------------------------
// Flash attention, bf16 WMMA, online softmax.
// grid = (SEQ/128, BATCH*NHEAD), block = 256 (8 waves); each wave owns 16
// query rows. K tile via TDM, V tile transposed into LDS by waves 4-7.
// Per key block per wave: 4 wmma (QK^T) + 4 wmma (PV).
// ---------------------------------------------------------------------------
__global__ __launch_bounds__(256) void flash_attn_wmma(
    const bf16_t* __restrict__ Q, const bf16_t* __restrict__ Kin,
    const bf16_t* __restrict__ V, bf16_t* __restrict__ Oat) {
  __shared__ bf16_t Ks[32][HDIM + 8];     // [key][hd]  (TDM-padded rows)
  __shared__ bf16_t Vt[HDIM][32 + 8];     // [hd][key]
  __shared__ bf16_t Ps[8][16][32 + 8];    // per-wave P re-layout (C -> A)

  const int tid  = threadIdx.x;
  const int lane = tid & 31;
  const int wave = tid >> 5;
  const int ln    = lane & 15;
  const int half8 = (lane < 16) ? 0 : 8;
  const int bh   = blockIdx.y;
  const int bIdx = bh >> 4, h = bh & 15;
  const size_t base = (size_t)bh * SEQ * HDIM;
  const bf16_t* Qb = Q + base;
  const bf16_t* Kb = Kin + base;
  const bf16_t* Vb = V + base;
  const int q0 = blockIdx.x * 128 + wave * 16;

  v16bf qa[2];
  {
    const bf16_t* qp = Qb + (size_t)(q0 + ln) * HDIM + half8;
    qa[0] = make_frag(qp, 16);
    qa[1] = make_frag(qp + 32, 16);
  }

  v8f o[4];
  float mrow[8], lsum[8];
#pragma unroll
  for (int j = 0; j < 4; ++j)
#pragma unroll
    for (int r = 0; r < 8; ++r) o[j][r] = 0.0f;
#pragma unroll
  for (int r = 0; r < 8; ++r) { mrow[r] = -3.0e38f; lsum[r] = 0.0f; }

  for (int kb0 = 0; kb0 < SEQ; kb0 += 32) {
    // --- stage tiles ---
#if HAVE_TDM
    if (is_wave0())
      tdm_load_2d_bf16(lds_off(&Ks[0][0]), Kb + (size_t)kb0 * HDIM, HDIM, 32, HDIM);
#else
    if (tid < 128) {
      const int key = tid >> 2, c = (tid & 3) * 16;
      const bf16_t* src = Kb + (size_t)(kb0 + key) * HDIM + c;
      *(v8bf*)&Ks[key][c]     = *(const v8bf*)src;
      *(v8bf*)&Ks[key][c + 8] = *(const v8bf*)(src + 8);
    }
#endif
    if (tid >= 128) {
      const int s2 = tid - 128;
      const int key = s2 >> 2, c = (s2 & 3) * 16;
      const bf16_t* src = Vb + (size_t)(kb0 + key) * HDIM + c;
      v8bf a = *(const v8bf*)src, b2 = *(const v8bf*)(src + 8);
#pragma unroll
      for (int i = 0; i < 8; ++i) { Vt[c + i][key] = a[i]; Vt[c + 8 + i][key] = b2[i]; }
    }
#if HAVE_TDM
    if (is_wave0()) __builtin_amdgcn_s_wait_tensorcnt(0);
#endif
    __syncthreads();

    // --- S = Q K^T (16 x 32) ---
    v8f s0, s1;
#pragma unroll
    for (int r = 0; r < 8; ++r) { s0[r] = 0.0f; s1[r] = 0.0f; }
#pragma unroll
    for (int ks = 0; ks < 2; ++ks) {
      v16bf b0 = make_frag(&Ks[ln][ks * 32 + half8], 16);
      v16bf b1 = make_frag(&Ks[16 + ln][ks * 32 + half8], 16);
      s0 = WMMA_BF16(qa[ks], b0, s0);
      s1 = WMMA_BF16(qa[ks], b1, s1);
    }

    // --- online softmax (row = VGPR r + half8, values across 16 lanes) ---
    float bm[8], ps[8];
#pragma unroll
    for (int r = 0; r < 8; ++r) {
      s0[r] *= ATTN_SCALE; s1[r] *= ATTN_SCALE;
      bm[r] = fmaxf(s0[r], s1[r]);
    }
#pragma unroll
    for (int off = 1; off < 16; off <<= 1)
#pragma unroll
      for (int r = 0; r < 8; ++r) bm[r] = fmaxf(bm[r], __shfl_xor(bm[r], off, 32));
#pragma unroll
    for (int r = 0; r < 8; ++r) {
      const float mn   = fmaxf(mrow[r], bm[r]);
      const float corr = exp2f((mrow[r] - mn) * LOG2E);
      mrow[r] = mn;
      const float p0 = exp2f((s0[r] - mn) * LOG2E);
      const float p1 = exp2f((s1[r] - mn) * LOG2E);
      s0[r] = p0; s1[r] = p1;
      ps[r] = p0 + p1;
#pragma unroll
      for (int j = 0; j < 4; ++j) o[j][r] *= corr;
      lsum[r] *= corr;
    }
#pragma unroll
    for (int off = 1; off < 16; off <<= 1)
#pragma unroll
      for (int r = 0; r < 8; ++r) ps[r] += __shfl_xor(ps[r], off, 32);
#pragma unroll
    for (int r = 0; r < 8; ++r) lsum[r] += ps[r];

    // --- P: C layout -> A layout via per-wave LDS bounce ---
#pragma unroll
    for (int r = 0; r < 8; ++r) {
      const int row = r + half8;
      Ps[wave][row][ln]      = (bf16_t)s0[r];
      Ps[wave][row][16 + ln] = (bf16_t)s1[r];
    }
    asm volatile("s_wait_dscnt 0" ::: "memory");  // DS in-order per wave
    v16bf pf = make_frag(&Ps[wave][ln][half8], 16);

    // --- O += P V ---
#pragma unroll
    for (int j = 0; j < 4; ++j) {
      v16bf bv = make_frag(&Vt[j * 16 + ln][half8], 16);
      o[j] = WMMA_BF16(pf, bv, o[j]);
    }
    __syncthreads();
  }

  // --- normalize, store bf16 [B*S, D], d = h*64 + hd ---
  float inv[8];
#pragma unroll
  for (int r = 0; r < 8; ++r) inv[r] = 1.0f / lsum[r];
  const size_t orow = (size_t)bIdx * SEQ + q0 + half8;
#pragma unroll
  for (int j = 0; j < 4; ++j) {
    const int d = h * HDIM + j * 16 + ln;
#pragma unroll
    for (int r = 0; r < 8; ++r)
      Oat[(orow + r) * DMODEL + d] = (bf16_t)(o[j][r] * inv[r]);
  }
}

// ---------------------------------------------------------------------------
extern "C" void kernel_launch(void* const* d_in, const int* in_sizes, int n_in,
                              void* d_out, int out_size, void* d_ws, size_t ws_size,
                              hipStream_t stream) {
  (void)in_sizes; (void)n_in; (void)out_size; (void)ws_size;
  const float* x      = (const float*)d_in[0];
  const float* qkv_w  = (const float*)d_in[1];
  const float* qkv_b  = (const float*)d_in[2];
  const float* proj_w = (const float*)d_in[3];
  const float* proj_b = (const float*)d_in[4];
  float* out = (float*)d_out;

  char* ws = (char*)d_ws;
  size_t off = 0;
  auto carve = [&](size_t bytes) {
    char* p = ws + off;
    off += (bytes + 255) & ~(size_t)255;
    return p;
  };
  const size_t MD = (size_t)BATCH * SEQ * DMODEL;
  bf16_t* xbf   = (bf16_t*)carve(MD * 2);
  bf16_t* wqkv  = (bf16_t*)carve(3 * (size_t)DMODEL * DMODEL * 2);
  bf16_t* wproj = (bf16_t*)carve((size_t)DMODEL * DMODEL * 2);
  bf16_t* qb    = (bf16_t*)carve(MD * 2);
  bf16_t* kb    = (bf16_t*)carve(MD * 2);
  bf16_t* vb    = (bf16_t*)carve(MD * 2);
  bf16_t* attn  = (bf16_t*)carve(MD * 2);

  cvt_f32_to_bf16<<<(int)(MD / 2048), 256, 0, stream>>>(x, xbf, (int)MD);
  cvt_f32_to_bf16<<<(3 * DMODEL * DMODEL) / 2048, 256, 0, stream>>>(qkv_w, wqkv, 3 * DMODEL * DMODEL);
  cvt_f32_to_bf16<<<(DMODEL * DMODEL) / 2048, 256, 0, stream>>>(proj_w, wproj, DMODEL * DMODEL);

  {
    dim3 grid(BATCH * SEQ / 128, 3 * DMODEL / 64);
    gemm_bf16_wmma<1><<<grid, 256, 0, stream>>>(xbf, wqkv, qkv_b, nullptr,
                                                qb, kb, vb,
                                                BATCH * SEQ, 3 * DMODEL, DMODEL);
  }
  {
    dim3 grid(SEQ / 128, BATCH * NHEAD);
    flash_attn_wmma<<<grid, 256, 0, stream>>>(qb, kb, vb, attn);
  }
  {
    dim3 grid(BATCH * SEQ / 128, DMODEL / 64);
    gemm_bf16_wmma<0><<<grid, 256, 0, stream>>>(attn, wproj, proj_b, out,
                                                nullptr, nullptr, nullptr,
                                                BATCH * SEQ, DMODEL, DMODEL);
  }
}
